// CosSimLoss_4922032521889
// MI455X (gfx1250) — matrix operand order repaired and checked by reading, not compiled, not run on hardware
//
#include <hip/hip_runtime.h>
#include <math.h>

typedef __attribute__((ext_vector_type(16))) _Float16 v16h;
typedef __attribute__((ext_vector_type(8)))  float    v8f;
typedef __attribute__((ext_vector_type(4)))  int      v4i;

#define N_ROWS 8192
#define CDIM   512
#define SCALE  1.6487212707001282f   /* exp(0.5) */
#define ROWS_PER_WG 128
#define COL_CHUNKS 8
#define COLS_PER_CHUNK (N_ROWS / COL_CHUNKS)        /* 1024 */
#define COL_TILES_PER_CHUNK (COLS_PER_CHUNK / 16)   /* 64  */
#define KTILES (CDIM / 32)                          /* 16  */
#define LDS_ROWSTRIDE 520   /* halves: 512 + 8 pad -> lane stride = 4 banks */

// ---------------------------------------------------------------------------
// CDNA5 async global->LDS copy (tracked by ASYNCcnt), with inline-asm fallback
// ---------------------------------------------------------------------------
typedef __attribute__((address_space(3))) _Float16 lds_f16_t;

__device__ __forceinline__ void async_copy_b128(const _Float16* gsrc, _Float16* ldst) {
#if __has_builtin(__builtin_amdgcn_global_load_async_to_lds_b128)
    void* gp = (void*)gsrc;   // drop const in generic AS first
    void* lp = (void*)ldst;
    __builtin_amdgcn_global_load_async_to_lds_b128(
        (__attribute__((address_space(1))) v4i*)gp,
        (__attribute__((address_space(3))) v4i*)lp, 0, 0);
#else
    unsigned loff = (unsigned)(size_t)(lds_f16_t*)ldst;   // 32-bit LDS byte offset
    asm volatile("global_load_async_to_lds_b128 %0, %1, off"
                 :: "v"(loff), "v"(gsrc) : "memory");
#endif
}

__device__ __forceinline__ void wait_async0() {
#if __has_builtin(__builtin_amdgcn_s_wait_asynccnt)
    __builtin_amdgcn_s_wait_asynccnt(0);
#else
    asm volatile("s_wait_asynccnt 0x0" ::: "memory");
#endif
}

// ---------------------------------------------------------------------------
// Kernel 1: per-row L2 normalize pred & target -> f16, and compute scaled diag
// ---------------------------------------------------------------------------
__global__ __launch_bounds__(256) void nrm_kernel(
    const float* __restrict__ pred, const float* __restrict__ tgt,
    _Float16* __restrict__ Pn, _Float16* __restrict__ Tn,
    float* __restrict__ diag)
{
    const int r = blockIdx.x;
    const int t = threadIdx.x;
    const float* pr = pred + (size_t)r * CDIM;
    const float* tr = tgt  + (size_t)r * CDIM;

    float p0 = pr[t], p1 = pr[t + 256];
    float q0 = tr[t], q1 = tr[t + 256];
    float ssp = p0 * p0 + p1 * p1;
    float sst = q0 * q0 + q1 * q1;
    float spt = p0 * q0 + p1 * q1;

    #pragma unroll
    for (int off = 16; off; off >>= 1) {
        ssp += __shfl_xor(ssp, off, 32);
        sst += __shfl_xor(sst, off, 32);
        spt += __shfl_xor(spt, off, 32);
    }
    __shared__ float red[3][8];
    const int wave = t >> 5;
    if ((t & 31) == 0) { red[0][wave] = ssp; red[1][wave] = sst; red[2][wave] = spt; }
    __syncthreads();
    float tssp = 0.f, tsst = 0.f, tspt = 0.f;
    #pragma unroll
    for (int w = 0; w < 8; ++w) { tssp += red[0][w]; tsst += red[1][w]; tspt += red[2][w]; }

    const float sp = 1.0f / fmaxf(sqrtf(tssp), 1e-12f);
    const float st = 1.0f / fmaxf(sqrtf(tsst), 1e-12f);

    Pn[(size_t)r * CDIM + t]       = (_Float16)(p0 * sp);
    Pn[(size_t)r * CDIM + t + 256] = (_Float16)(p1 * sp);
    Tn[(size_t)r * CDIM + t]       = (_Float16)(q0 * st);
    Tn[(size_t)r * CDIM + t + 256] = (_Float16)(q1 * st);
    if (t == 0) diag[r] = tspt * sp * st * SCALE;
}

// ---------------------------------------------------------------------------
// Kernel 2: fused GEMM (P x T^T) + exp + per-row sum.
// 8 waves/WG; each wave owns 16 rows, full 16x512 f16 A panel in registers.
// B column tiles (16 KB) are copied global->LDS with ASYNC loads, double-
// buffered (fence = s_wait_asynccnt 0 + barrier). Inside a tile, B k-steps
// are consumed in batches of 4 distinct register fragments so 8 ds_load_b128
// issue back-to-back and their latency amortizes over 4 WMMAs.
// ---------------------------------------------------------------------------
__global__ __launch_bounds__(256) void gemm_lse_kernel(
    const _Float16* __restrict__ Pn, const _Float16* __restrict__ Tn,
    float* __restrict__ partial)
{
    __shared__ _Float16 bbuf[2][16 * LDS_ROWSTRIDE];

    const int wave = threadIdx.x >> 5;
    const int lane = threadIdx.x & 31;
    const int m    = lane & 15;   // row (A) / column (B) within the 16-tile
    const int hi   = lane >> 4;   // lane-half selects the K sub-strip
    const int r0     = blockIdx.x * ROWS_PER_WG + wave * 16;
    const int c0base = blockIdx.y * COLS_PER_CHUNK;

    // Cooperative B-loader mapping: 16 threads per column, 64 B per thread.
    const int lc  = threadIdx.x >> 4;   // 0..15: local column
    const int sub = threadIdx.x & 15;   // 0..15: 32-half (64 B) segment

    // --- Load A panel: 16 rows x 512 K, WMMA 16-bit A layout -------------
    const _Float16* arow = Pn + (size_t)(r0 + m) * CDIM;
    v16h a[KTILES];
    #pragma unroll
    for (int kt = 0; kt < KTILES; ++kt) {
        const int koff = kt * 32 + hi * 8;
        union { v16h v; float4 f[2]; } u;
        u.f[0] = *reinterpret_cast<const float4*>(arow + koff);
        u.f[1] = *reinterpret_cast<const float4*>(arow + koff + 16);
        a[kt] = u.v;
    }

    float s[8];
    #pragma unroll
    for (int r = 0; r < 8; ++r) s[r] = 0.f;

    // Per-thread async-copy source/dest: 64 B of one column tile, 4 x B128.
    const _Float16* gtile0 = Tn + (size_t)(c0base + lc) * CDIM + sub * 32;
    const size_t gtile_step = (size_t)16 * CDIM;               // halves per tile
    _Float16* const lslot0 = &bbuf[0][lc * LDS_ROWSTRIDE + sub * 32];
    _Float16* const lslot1 = &bbuf[1][lc * LDS_ROWSTRIDE + sub * 32];

    // ---- prologue: async-fetch tile 0 into buf 0 ------------------------
    {
        const _Float16* src = gtile0;
        async_copy_b128(src,      lslot0);
        async_copy_b128(src + 8,  lslot0 + 8);
        async_copy_b128(src + 16, lslot0 + 16);
        async_copy_b128(src + 24, lslot0 + 24);
    }
    wait_async0();
    __syncthreads();

    #pragma unroll 1
    for (int jt = 0; jt < COL_TILES_PER_CHUNK; ++jt) {
        // Issue async copies for the NEXT tile (runs behind the WMMAs).
        if (jt + 1 < COL_TILES_PER_CHUNK) {
            const _Float16* src = gtile0 + (size_t)(jt + 1) * gtile_step;
            _Float16* dst = ((jt + 1) & 1) ? lslot1 : lslot0;
            async_copy_b128(src,      dst);
            async_copy_b128(src + 8,  dst + 8);
            async_copy_b128(src + 16, dst + 16);
            async_copy_b128(src + 24, dst + 24);
        }

        // Compute on current LDS buffer; consume B in batches of 4 k-tiles.
        const _Float16* brow = &bbuf[jt & 1][m * LDS_ROWSTRIDE + hi * 8];
        union BU { v16h v; float4 f[2]; };

        v8f c = {0.f, 0.f, 0.f, 0.f, 0.f, 0.f, 0.f, 0.f};
        #pragma unroll
        for (int kb = 0; kb < KTILES; kb += 4) {
            BU b0, b1, b2, b3;
            b0.f[0] = *reinterpret_cast<const float4*>(brow + (kb + 0) * 32);
            b0.f[1] = *reinterpret_cast<const float4*>(brow + (kb + 0) * 32 + 16);
            b1.f[0] = *reinterpret_cast<const float4*>(brow + (kb + 1) * 32);
            b1.f[1] = *reinterpret_cast<const float4*>(brow + (kb + 1) * 32 + 16);
            b2.f[0] = *reinterpret_cast<const float4*>(brow + (kb + 2) * 32);
            b2.f[1] = *reinterpret_cast<const float4*>(brow + (kb + 2) * 32 + 16);
            b3.f[0] = *reinterpret_cast<const float4*>(brow + (kb + 3) * 32);
            b3.f[1] = *reinterpret_cast<const float4*>(brow + (kb + 3) * 32 + 16);
            c = __builtin_amdgcn_wmma_f32_16x16x32_f16(
                    false, a[kb + 0], false, b0.v, (short)0, c, false, false);
            c = __builtin_amdgcn_wmma_f32_16x16x32_f16(
                    false, a[kb + 1], false, b1.v, (short)0, c, false, false);
            c = __builtin_amdgcn_wmma_f32_16x16x32_f16(
                    false, a[kb + 2], false, b2.v, (short)0, c, false, false);
            c = __builtin_amdgcn_wmma_f32_16x16x32_f16(
                    false, a[kb + 3], false, b3.v, (short)0, c, false, false);
        }

        // |dot| <= 1 -> scale*dot in [-1.65, 1.65]: no max-subtraction needed
        #pragma unroll
        for (int r = 0; r < 8; ++r) s[r] += __expf(SCALE * c[r]);

        // Fence the async tile and make it visible to all waves.
        if (jt + 1 < COL_TILES_PER_CHUNK) {
            wait_async0();
            __syncthreads();
        }
    }

    // C layout: VGPR r, lane L -> M = r + 8*(L>=16), N = L&15.
    #pragma unroll
    for (int r = 0; r < 8; ++r) {
        float v = s[r];
        v += __shfl_xor(v, 8, 16);
        v += __shfl_xor(v, 4, 16);
        v += __shfl_xor(v, 2, 16);
        v += __shfl_xor(v, 1, 16);
        if (m == 0) {
            const int row = r0 + r + hi * 8;
            partial[(size_t)blockIdx.y * N_ROWS + row] = v;  // own slot, no atomics
        }
    }
}

// ---------------------------------------------------------------------------
// Kernel 3: loss = mean( log(sum_chunks partial[row]) - diag[row] )
// ---------------------------------------------------------------------------
__global__ __launch_bounds__(256) void finish_kernel(
    const float* __restrict__ partial, const float* __restrict__ diag,
    float* __restrict__ out)
{
    const int t = threadIdx.x;
    float acc = 0.f;
    for (int i = t; i < N_ROWS; i += 256) {
        float e = 0.f;
        #pragma unroll
        for (int ch = 0; ch < COL_CHUNKS; ++ch) e += partial[(size_t)ch * N_ROWS + i];
        acc += logf(e) - diag[i];
    }
    #pragma unroll
    for (int off = 16; off; off >>= 1) acc += __shfl_xor(acc, off, 32);
    __shared__ float red[8];
    if ((t & 31) == 0) red[t >> 5] = acc;
    __syncthreads();
    if (t == 0) {
        float tot = 0.f;
        #pragma unroll
        for (int w = 0; w < 8; ++w) tot += red[w];
        out[0] = tot * (1.0f / (float)N_ROWS);
    }
}

// ---------------------------------------------------------------------------
extern "C" void kernel_launch(void* const* d_in, const int* in_sizes, int n_in,
                              void* d_out, int out_size, void* d_ws, size_t ws_size,
                              hipStream_t stream) {
    const float* pred = (const float*)d_in[0];
    const float* tgt  = (const float*)d_in[1];

    char* ws = (char*)d_ws;
    _Float16* Pn   = (_Float16*)ws;                                   // 8 MiB
    _Float16* Tn   = (_Float16*)(ws + (size_t)N_ROWS * CDIM * 2);     // 8 MiB
    float*    diag = (float*)(ws + (size_t)N_ROWS * CDIM * 4);        // 32 KiB
    float*    part = diag + N_ROWS;                                   // 256 KiB

    nrm_kernel<<<N_ROWS, 256, 0, stream>>>(pred, tgt, Pn, Tn, diag);

    dim3 grid(N_ROWS / ROWS_PER_WG, COL_CHUNKS);   // (64, 8)
    gemm_lse_kernel<<<grid, 256, 0, stream>>>(Pn, Tn, part);

    finish_kernel<<<1, 256, 0, stream>>>(part, diag, (float*)d_out);
}
